// RNN_VANILLA_10694468567412
// MI455X (gfx1250) — matrix operand order repaired and checked by reading, not compiled, not run on hardware
//
#include <hip/hip_runtime.h>

// ---------------- problem constants ----------------
constexpr int kT = 1024;
constexpr int kB = 512;
constexpr int kI = 32;
constexpr int kH = 256;
constexpr int kO = 32;
constexpr int kG = 4 * kH;   // 1024 gate columns

// ---------------- WMMA types ----------------
typedef float  v8f   __attribute__((ext_vector_type(8)));
typedef __bf16 v16bf __attribute__((ext_vector_type(16)));

// explicit global-address-space pointers: force global_load/store (not FLAT,
// which would tie up the LDS pipe and couple LOADcnt with DScnt)
typedef const v16bf __attribute__((address_space(1)))*  GWf;
typedef const float __attribute__((address_space(1)))*  GCF;
typedef float       __attribute__((address_space(1)))*  GF;

union AFrag {
    v16bf v;
    uint4 q[2];
    unsigned short s[16];
};

__device__ __forceinline__ unsigned short f2bf(float f) {
    union { float f; unsigned u; } cv; cv.f = f;
    unsigned u = cv.u;
    return (unsigned short)((u + 0x7FFFu + ((u >> 16) & 1u)) >> 16); // RNE
}

// ---------------- pre-swizzled weight workspace layout ----------------
// Fragment-major: [frag][lane(32)][16 bf16] — one 32B v16bf per lane per fragment.
constexpr int FRAG_HH  = 64 * 8 * 32;  // 64 N-tiles x 8 K-chunks x 32 lanes
constexpr int FRAG_IH  = 64 * 1 * 32;  // 64 N-tiles x 1 K-chunk
constexpr int FRAG_OUT = 2 * 8 * 32;   // 2 N-tiles x 8 K-chunks
constexpr int VOFF_IH  = FRAG_HH;             // in v16bf units
constexpr int VOFF_OUT = FRAG_HH + FRAG_IH;   // in v16bf units
constexpr int FRAG_TOTAL = FRAG_HH + FRAG_IH + FRAG_OUT;

// B-fragment layout (16-bit, 32x16): lane l -> n = l&15, Kgroup = l>>4;
// element e -> K = 8*(l>>4) + (e&7) + 16*(e>>3). B(k,n) = W[n][k]  (we compute h @ W^T).
__global__ void lstm_swizzle(const float* __restrict__ Wih,
                             const float* __restrict__ Whh,
                             const float* __restrict__ Wout,
                             unsigned short* __restrict__ ws) {
    int gid = blockIdx.x * blockDim.x + threadIdx.x;
    const float* src;
    if (gid < FRAG_HH) {
        int l = gid & 31, kk = (gid >> 5) & 7, n0 = gid >> 8;
        src = Whh + (size_t)(n0 * 16 + (l & 15)) * kH + kk * 32 + 8 * (l >> 4);
    } else if (gid < FRAG_HH + FRAG_IH) {
        int g = gid - FRAG_HH;
        int l = g & 31, n0 = g >> 5;
        src = Wih + (size_t)(n0 * 16 + (l & 15)) * kI + 8 * (l >> 4);
    } else if (gid < FRAG_TOTAL) {
        int g = gid - FRAG_HH - FRAG_IH;
        int l = g & 31, kk = (g >> 5) & 7, ot = g >> 8;
        src = Wout + (size_t)(ot * 16 + (l & 15)) * kH + kk * 32 + 8 * (l >> 4);
    } else {
        return;
    }
    unsigned short* dst = ws + (size_t)gid * 16;
#pragma unroll
    for (int e = 0; e < 16; ++e) dst[e] = f2bf(src[(e & 7) + ((e >> 3) << 4)]);
}

// ---------------- dynamic LDS layout ----------------
constexpr int HPAD = 264;   // bf16 elems per h row (16B-aligned, bank-spread)
constexpr int GPAD = 1032;  // fp32 elems per gates row
constexpr int LDS_HBUF  = 16 * GPAD * 4;            // gates: 66048B at offset 0
constexpr int LDS_CBUF  = LDS_HBUF + 16 * HPAD * 2; // h bf16: 8448B
constexpr int LDS_BIAS  = LDS_CBUF + 16 * kH * 4;   // c fp32: 16384B
constexpr int LDS_BOUT  = LDS_BIAS + kG * 4;        // bias:   4096B
constexpr int LDS_TOTAL = LDS_BOUT + kO * 4;        // 95104B total

constexpr int NTHREADS = 512;  // 16 waves -> 4 waves/SIMD -> ~256 VGPRs/wave

__global__ __launch_bounds__(NTHREADS)
void lstm_persistent(const float* __restrict__ xin_,   // [T,B,I]
                     const float* __restrict__ phys_,  // [T,B,O]
                     const float* __restrict__ bih,
                     const float* __restrict__ bhh,
                     const float* __restrict__ bout,
                     const unsigned short* __restrict__ ws,
                     float* __restrict__ outFull_,     // [B,T,O]
                     float* __restrict__ outRnn_) {    // [B,T,O]
    extern __shared__ char smem[];
    float*          gbuf = (float*)smem;
    unsigned short* hbuf = (unsigned short*)(smem + LDS_HBUF);
    float*          cbuf = (float*)(smem + LDS_CBUF);
    float*          bias = (float*)(smem + LDS_BIAS);
    float*          bo   = (float*)(smem + LDS_BOUT);

    GCF xin     = (GCF)xin_;
    GCF phys    = (GCF)phys_;
    GF  outFull = (GF)outFull_;
    GF  outRnn  = (GF)outRnn_;

    const int tid  = threadIdx.x;
    const int lane = tid & 31;
    const int wv   = tid >> 5;        // 16 waves
    const int nn   = lane & 15;       // A row / B col within tile
    const int half = lane >> 4;
    const int b0   = blockIdx.x * 16; // batch tile base
    const int col0 = wv * 64;         // this wave's gate-column range (4 subtiles)

    // init h=0, c=0, fold biases
    for (int i = tid; i < 16 * HPAD; i += NTHREADS) hbuf[i] = 0;
    for (int i = tid; i < 16 * kH;   i += NTHREADS) cbuf[i] = 0.f;
    bias[tid]       = bih[tid]       + bhh[tid];
    bias[tid + 512] = bih[tid + 512] + bhh[tid + 512];
    if (tid < kO) bo[tid] = bout[tid];
    __syncthreads();

    unsigned long long wsa = (unsigned long long)ws;

    for (int t = 0; t < kT; ++t) {
        // Defeat LICM: make the weight base pointer loop-variant so B-fragment
        // loads stay inside the loop (streamed from L2-resident ws) instead of
        // being hoisted into ~300 VGPRs and spilled to scratch. Cast to an
        // address_space(1) pointer so they lower to global_load_b128 (not FLAT).
        asm volatile("" : "+s"(wsa));
        GWf Wf = (GWf)wsa;

        // ---- phase 1: gates = h @ Whh^T + x @ Wih^T (bf16 WMMA, f32 acc)
        AFrag a[8];
#pragma unroll
        for (int kk = 0; kk < 8; ++kk) {
            // A layout: lane -> M = lane&15; elems 0..7 -> K = 8*half+e,
            //           elems 8..15 -> K = 16 + 8*half + (e-8)
            const uint4* p =
                (const uint4*)(hbuf + (nn * HPAD + kk * 32 + 8 * half));
            a[kk].q[0] = p[0];
            a[kk].q[1] = p[2];
        }
        AFrag ax;
        {
            GCF xrow = xin + ((size_t)t * kB + b0 + nn) * kI;
            const int kb = 8 * half;
#pragma unroll
            for (int e = 0; e < 8; ++e) ax.s[e]     = f2bf(xrow[kb + e]);
#pragma unroll
            for (int e = 0; e < 8; ++e) ax.s[8 + e] = f2bf(xrow[kb + 16 + e]);
        }
#pragma unroll
        for (int s = 0; s < 4; ++s) {
            v8f acc = {0.f, 0.f, 0.f, 0.f, 0.f, 0.f, 0.f, 0.f};
            const int nT = wv * 4 + s;
#pragma unroll
            for (int kk = 0; kk < 8; ++kk) {
                v16bf bb = Wf[(nT * 8 + kk) * 32 + lane];  // streamed from L2
                acc = __builtin_amdgcn_wmma_f32_16x16x32_bf16(
                    false, a[kk].v, false, bb, (short)0, acc, false, false);
            }
            {
                v16bf bb = Wf[VOFF_IH + nT * 32 + lane];
                acc = __builtin_amdgcn_wmma_f32_16x16x32_bf16(
                    false, ax.v, false, bb, (short)0, acc, false, false);
            }
            // C/D layout: VGPR r, lane -> (M = r + 8*half, N = lane&15)
#pragma unroll
            for (int r = 0; r < 8; ++r)
                gbuf[(r + 8 * half) * GPAD + col0 + s * 16 + nn] = acc[r];
        }
        __syncthreads();

        // ---- phase 2: pointwise LSTM cell (fp32)
#pragma unroll
        for (int k = 0; k < 8; ++k) {
            int idx = tid + k * NTHREADS;       // 16*256 = 4096 cells
            int m = idx >> 8, hid = idx & 255;
            const float* gr = gbuf + m * GPAD + hid;
            float ig = gr[0]   + bias[hid];
            float fg = gr[256] + bias[256 + hid];
            float gg = gr[512] + bias[512 + hid];
            float og = gr[768] + bias[768 + hid];
            ig = 1.f / (1.f + __expf(-ig));
            fg = 1.f / (1.f + __expf(-fg));
            og = 1.f / (1.f + __expf(-og));
            gg = tanhf(gg);
            float c = fg * cbuf[m * kH + hid] + ig * gg;
            cbuf[m * kH + hid] = c;
            float h = og * tanhf(c);
            hbuf[m * HPAD + hid] = f2bf(h);
        }
        __syncthreads();

        // ---- phase 3: output head on waves 0,1: rnn = h @ Wout^T + b_out
        if (wv < 2) {
            v8f acc = {0.f, 0.f, 0.f, 0.f, 0.f, 0.f, 0.f, 0.f};
#pragma unroll
            for (int kk = 0; kk < 8; ++kk) {
                AFrag av;
                const uint4* p =
                    (const uint4*)(hbuf + (nn * HPAD + kk * 32 + 8 * half));
                av.q[0] = p[0];
                av.q[1] = p[2];
                v16bf bb = Wf[VOFF_OUT + (wv * 8 + kk) * 32 + lane];
                acc = __builtin_amdgcn_wmma_f32_16x16x32_bf16(
                    false, av.v, false, bb, (short)0, acc, false, false);
            }
            const int o   = wv * 16 + nn;
            const float bov = bo[o];
#pragma unroll
            for (int r = 0; r < 8; ++r) {
                int   bIdx = b0 + r + 8 * half;
                float rnn  = acc[r] + bov;
                float ph   = phys[((size_t)t * kB + bIdx) * kO + o];
                size_t oi  = (size_t)bIdx * (kT * kO) + (size_t)t * kO + o;
                outFull[oi] = ph + rnn;
                outRnn[oi]  = rnn;
            }
        }
        __syncthreads();
    }
}

extern "C" void kernel_launch(void* const* d_in, const int* in_sizes, int n_in,
                              void* d_out, int out_size, void* d_ws, size_t ws_size,
                              hipStream_t stream) {
    const float* xin  = (const float*)d_in[0];
    const float* phys = (const float*)d_in[1];
    const float* Wih  = (const float*)d_in[2];
    const float* Whh  = (const float*)d_in[3];
    const float* bih  = (const float*)d_in[4];
    const float* bhh  = (const float*)d_in[5];
    const float* Wout = (const float*)d_in[6];
    const float* bout = (const float*)d_in[7];

    unsigned short* ws = (unsigned short*)d_ws;   // needs ~593 KB
    float* outFull = (float*)d_out;
    float* outRnn  = outFull + (size_t)kB * kT * kO;

    // 1) swizzle weights into WMMA-fragment-major bf16 (runs every call; deterministic)
    lstm_swizzle<<<(FRAG_TOTAL + 255) / 256, 256, 0, stream>>>(Wih, Whh, Wout, ws);

    // 2) persistent per-batch-tile LSTM rollout: 32 WGs x 512 threads (16 waves)
    lstm_persistent<<<kB / 16, NTHREADS, LDS_TOTAL, stream>>>(
        xin, phys, bih, bhh, bout, ws, outFull, outRnn);
}
